// orchestra_68667937129258
// MI455X (gfx1250) — matrix operand order repaired and checked by reading, not compiled, not run on hardware
//
#include <hip/hip_runtime.h>
#include <hip/hip_bf16.h>
#include <stdint.h>

#define N_ROWS 524288
#define K_DIM  128
constexpr float LAMD  = 25.0f;
constexpr float INV_K = 1.0f / 128.0f;
constexpr float INV_N = 1.0f / 524288.0f;
constexpr float TOL   = 0.01f;

typedef __bf16 v16bf __attribute__((ext_vector_type(16)));
typedef __bf16 v8bf  __attribute__((ext_vector_type(8)));
typedef float  v8f   __attribute__((ext_vector_type(8)));

__device__ __forceinline__ uint16_t f2bf(float f) {
  uint32_t u = __float_as_uint(f);
  u += 0x7FFFu + ((u >> 16) & 1u);   // round-to-nearest-even
  return (uint16_t)(u >> 16);
}
__device__ __forceinline__ float bflo(uint32_t u) { return __uint_as_float(u << 16); }
__device__ __forceinline__ float bfhi(uint32_t u) { return __uint_as_float(u & 0xFFFF0000u); }

// ---------------------------------------------------------------------------
// Setup: probs = softmax(cZ*lamd, axis=1) stored as bf16 [N,K]; c = 1/N.
// One wave (32 lanes) per row; lane owns 4 contiguous f32 (float4 load).
// ---------------------------------------------------------------------------
__global__ __launch_bounds__(256) void softmax_kernel(
    const float* __restrict__ cZ, uint16_t* __restrict__ P, float* __restrict__ c) {
  int wave = (int)((blockIdx.x * blockDim.x + threadIdx.x) >> 5);
  int lane = threadIdx.x & 31;
  if (wave >= N_ROWS) return;
  const float4 x4 = *(const float4*)(cZ + (size_t)wave * K_DIM + lane * 4);
  float a0 = x4.x * LAMD, a1 = x4.y * LAMD, a2 = x4.z * LAMD, a3 = x4.w * LAMD;
  float m = fmaxf(fmaxf(a0, a1), fmaxf(a2, a3));
  for (int off = 16; off; off >>= 1) m = fmaxf(m, __shfl_xor(m, off, 32));
  float e0 = __expf(a0 - m), e1 = __expf(a1 - m), e2 = __expf(a2 - m), e3 = __expf(a3 - m);
  float s = e0 + e1 + e2 + e3;
  for (int off = 16; off; off >>= 1) s += __shfl_xor(s, off, 32);
  float inv = 1.0f / s;
  uint32_t w0 = (uint32_t)f2bf(e0 * inv) | ((uint32_t)f2bf(e1 * inv) << 16);
  uint32_t w1 = (uint32_t)f2bf(e2 * inv) | ((uint32_t)f2bf(e3 * inv) << 16);
  uint2 pk = make_uint2(w0, w1);
  *(uint2*)(P + (size_t)wave * K_DIM + lane * 4) = pk;
  if (lane == 0) c[wave] = INV_N;
}

__global__ void init_small_kernel(float* r_accum, float* err_accum, int* stop) {
  int t = threadIdx.x;
  if (t < K_DIM) r_accum[t] = 0.0f;
  if (t == 0) { *err_accum = 0.0f; *stop = 0; }
}

// ---------------------------------------------------------------------------
// r_accum[k] += sum_j P[j,k] * c[j].  Streaming, coalesced (wave reads one
// contiguous 256B row slice), LDS cross-row reduce, 128 atomics per block.
// ---------------------------------------------------------------------------
__global__ __launch_bounds__(256) void r_partial_kernel(
    const uint16_t* __restrict__ P, const float* __restrict__ c,
    float* __restrict__ r_accum, const int* __restrict__ stop) {
  if (*stop) return;
  const int kg = threadIdx.x & 31;   // 4 consecutive k per thread
  const int js = threadIdx.x >> 5;   // row slot within block (8 rows in flight)
  float acc0 = 0.f, acc1 = 0.f, acc2 = 0.f, acc3 = 0.f;
  const int base = blockIdx.x * 1024;
#pragma unroll 4
  for (int s = 0; s < 128; ++s) {
    int j = base + js + 8 * s;
    float cj = c[j];
    const uint16_t* pr = P + (size_t)j * K_DIM + kg * 4;
    __builtin_prefetch(pr + 32 * K_DIM, 0, 0);  // global_prefetch_b8, 32 rows ahead
    uint2 p = *(const uint2*)pr;
    acc0 = fmaf(cj, bflo(p.x), acc0);
    acc1 = fmaf(cj, bfhi(p.x), acc1);
    acc2 = fmaf(cj, bflo(p.y), acc2);
    acc3 = fmaf(cj, bfhi(p.y), acc3);
  }
  __shared__ float red[1024];
  red[js * 128 + kg * 4 + 0] = acc0;
  red[js * 128 + kg * 4 + 1] = acc1;
  red[js * 128 + kg * 4 + 2] = acc2;
  red[js * 128 + kg * 4 + 3] = acc3;
  __syncthreads();
  if (threadIdx.x < 128) {
    float t = 0.f;
#pragma unroll
    for (int q = 0; q < 8; ++q) t += red[q * 128 + threadIdx.x];
    atomicAdd(&r_accum[threadIdx.x], t);
  }
}

// ---------------------------------------------------------------------------
// r[k] = inv_K / r_accum[k]; zero accum; pack r into WMMA B-operand layout:
// chunk cc (k = 32cc..32cc+31): r_packed[cc*16 + h*8 + v] = pack(r[2i], r[2i+1])
// with i = 16cc + 8h + v  (lane-half h gets K 16h..16h+15 as pairs per VGPR).
// ---------------------------------------------------------------------------
__global__ __launch_bounds__(128) void r_finalize_kernel(
    float* __restrict__ r, float* __restrict__ r_accum,
    uint32_t* __restrict__ r_packed, const int* __restrict__ stop) {
  if (*stop) return;
  int k = threadIdx.x;
  float rk = INV_K / r_accum[k];
  r[k] = rk;
  r_accum[k] = 0.0f;
  __shared__ float rs[K_DIM];
  rs[k] = rk;
  __syncthreads();
  if (k < 64)
    r_packed[k] = (uint32_t)f2bf(rs[2 * k]) | ((uint32_t)f2bf(rs[2 * k + 1]) << 16);
}

// ---------------------------------------------------------------------------
// c[i] = inv_N / sum_k P[i,k]*r[k]  via v_wmma_f32_16x16x32_bf16.
// A = 16x32 bf16 tile of P (lane&15 = row M, lane>>4 selects K-half 16h..16h+15 in
// contiguous order, same as B). B = r replicated across 16 columns, loaded once per wave
// and reused for 8 row-tiles. All 8 A-loads for a tile are issued as one
// clause and pinned before the 4 chained WMMAs with a sched_barrier, so the
// wave keeps 8 b128 loads in flight per s_wait (L2-bandwidth-bound loop).
// D columns are all equal => lanes 0 and 16 each own 8 rows (M = v + 8h) and
// publish c with two b128 stores. Every 10th iteration:
// err += nansum(|c_old*acc*N - 1|) == |c_old/c_new - 1| (division-free).
// ---------------------------------------------------------------------------
__global__ __launch_bounds__(256) void c_update_kernel(
    const uint16_t* __restrict__ P, float* __restrict__ c,
    const uint32_t* __restrict__ r_packed, float* __restrict__ err_accum,
    const int* __restrict__ stop, int check_err) {
  if (*stop) return;
  const int lane = threadIdx.x & 31;
  const int wave = (int)((blockIdx.x * blockDim.x + threadIdx.x) >> 5);
  const int h = lane >> 4;
  const int mrow = lane & 15;

  const v16bf b0 = *(const v16bf*)(r_packed + 0 * 16 + h * 8);
  const v16bf b1 = *(const v16bf*)(r_packed + 1 * 16 + h * 8);
  const v16bf b2 = *(const v16bf*)(r_packed + 2 * 16 + h * 8);
  const v16bf b3 = *(const v16bf*)(r_packed + 3 * 16 + h * 8);

  float err_local = 0.0f;
  const int TPW = 8;  // tiles (16 rows each) per wave
#pragma unroll 1
  for (int t = 0; t < TPW; ++t) {
    const int i0 = (wave * TPW + t) * 16;
    const uint16_t* pA = P + (size_t)(i0 + mrow) * K_DIM + h * 16;
    // Issue ALL 8 b128 loads up front; sched_barrier pins them before the
    // WMMA group so the backend cannot re-sink them into load/wait pairs.
    v8bf l0 = *(const v8bf*)(pA + 0);
    v8bf h0 = *(const v8bf*)(pA + 8);
    v8bf l1 = *(const v8bf*)(pA + 32);
    v8bf h1 = *(const v8bf*)(pA + 40);
    v8bf l2 = *(const v8bf*)(pA + 64);
    v8bf h2 = *(const v8bf*)(pA + 72);
    v8bf l3 = *(const v8bf*)(pA + 96);
    v8bf h3 = *(const v8bf*)(pA + 104);
    __builtin_amdgcn_sched_barrier(0);
    v16bf a0 = __builtin_shufflevector(l0, h0, 0,1,2,3,4,5,6,7,8,9,10,11,12,13,14,15);
    v16bf a1 = __builtin_shufflevector(l1, h1, 0,1,2,3,4,5,6,7,8,9,10,11,12,13,14,15);
    v16bf a2 = __builtin_shufflevector(l2, h2, 0,1,2,3,4,5,6,7,8,9,10,11,12,13,14,15);
    v16bf a3 = __builtin_shufflevector(l3, h3, 0,1,2,3,4,5,6,7,8,9,10,11,12,13,14,15);
    v8f acc = {};
    acc = __builtin_amdgcn_wmma_f32_16x16x32_bf16(false, a0, false, b0, (short)0, acc, false, false);
    acc = __builtin_amdgcn_wmma_f32_16x16x32_bf16(false, a1, false, b1, (short)0, acc, false, false);
    acc = __builtin_amdgcn_wmma_f32_16x16x32_bf16(false, a2, false, b2, (short)0, acc, false, false);
    acc = __builtin_amdgcn_wmma_f32_16x16x32_bf16(false, a3, false, b3, (short)0, acc, false, false);

    if (mrow == 0) {  // lanes 0 and 16 own M = 0..7 and 8..15
      const int idx0 = i0 + h * 8;
      if (check_err) {
        // Old c loaded as two float4 (one wait), terms are division-free.
        float4 co0 = *(const float4*)(c + idx0);
        float4 co1 = *(const float4*)(c + idx0 + 4);
        const float nn = (float)N_ROWS;
        float eo[8] = {
          fabsf(fmaf(co0.x * nn, acc[0], -1.0f)), fabsf(fmaf(co0.y * nn, acc[1], -1.0f)),
          fabsf(fmaf(co0.z * nn, acc[2], -1.0f)), fabsf(fmaf(co0.w * nn, acc[3], -1.0f)),
          fabsf(fmaf(co1.x * nn, acc[4], -1.0f)), fabsf(fmaf(co1.y * nn, acc[5], -1.0f)),
          fabsf(fmaf(co1.z * nn, acc[6], -1.0f)), fabsf(fmaf(co1.w * nn, acc[7], -1.0f))};
#pragma unroll
        for (int v = 0; v < 8; ++v)
          if (!(eo[v] != eo[v])) err_local += eo[v];  // nansum: drop NaN terms
      }
      float4 cn0, cn1;
      cn0.x = INV_N * __builtin_amdgcn_rcpf(acc[0]);
      cn0.y = INV_N * __builtin_amdgcn_rcpf(acc[1]);
      cn0.z = INV_N * __builtin_amdgcn_rcpf(acc[2]);
      cn0.w = INV_N * __builtin_amdgcn_rcpf(acc[3]);
      cn1.x = INV_N * __builtin_amdgcn_rcpf(acc[4]);
      cn1.y = INV_N * __builtin_amdgcn_rcpf(acc[5]);
      cn1.z = INV_N * __builtin_amdgcn_rcpf(acc[6]);
      cn1.w = INV_N * __builtin_amdgcn_rcpf(acc[7]);
      *(float4*)(c + idx0) = cn0;       // two b128 stores, 32B-aligned
      *(float4*)(c + idx0 + 4) = cn1;
    }
  }
  if (check_err && mrow == 0 && err_local != 0.0f) atomicAdd(err_accum, err_local);
}

__global__ void err_finalize_kernel(float* err_accum, int* stop) {
  if (*stop) return;
  if (*err_accum < TOL) *stop = 1;
  *err_accum = 0.0f;
}

// ---------------------------------------------------------------------------
// out[i,k] = P[i,k] * c[i] * r[k] * N   (f32, float4-coalesced)
// ---------------------------------------------------------------------------
__global__ __launch_bounds__(256) void output_kernel(
    const uint16_t* __restrict__ P, const float* __restrict__ c,
    const float* __restrict__ r, float* __restrict__ out) {
  size_t e = ((size_t)blockIdx.x * blockDim.x + threadIdx.x) * 4;
  size_t row = e >> 7;
  int k = (int)(e & 127);
  uint2 p = *(const uint2*)(P + e);
  float ci = c[row] * (float)N_ROWS;
  float4 rv = *(const float4*)(r + k);
  float4 o;
  o.x = bflo(p.x) * ci * rv.x;
  o.y = bfhi(p.x) * ci * rv.y;
  o.z = bflo(p.y) * ci * rv.z;
  o.w = bfhi(p.y) * ci * rv.w;
  *(float4*)(out + e) = o;
}

extern "C" void kernel_launch(void* const* d_in, const int* in_sizes, int n_in,
                              void* d_out, int out_size, void* d_ws, size_t ws_size,
                              hipStream_t stream) {
  const float* cZ = (const float*)d_in[0];
  float* out = (float*)d_out;

  // Workspace layout (all offsets keep 32B alignment):
  uint16_t* P        = (uint16_t*)d_ws;                         // N*K bf16 = 128 MB
  float*    c        = (float*)(P + (size_t)N_ROWS * K_DIM);    // N f32 = 2 MB
  float*    r        = c + N_ROWS;                              // 128 f32
  float*    r_accum  = r + K_DIM;                               // 128 f32
  uint32_t* r_packed = (uint32_t*)(r_accum + K_DIM);            // 64 dwords
  float*    err_acc  = (float*)(r_packed + 64);
  int*      stop     = (int*)(err_acc + 8);

  // Setup: softmax -> bf16 probs (fits the 192MB L2), c0 = 1/N, zero state.
  softmax_kernel<<<N_ROWS / 8, 256, 0, stream>>>(cZ, P, c);
  init_small_kernel<<<1, 256, 0, stream>>>(r_accum, err_acc, stop);

  // Fixed 100-iteration schedule; device-side stop flag emulates while_loop.
  for (int it = 0; it < 100; ++it) {
    const int check = (it % 10 == 0) ? 1 : 0;
    r_partial_kernel<<<N_ROWS / 1024, 256, 0, stream>>>(P, c, r_accum, stop);
    r_finalize_kernel<<<1, 128, 0, stream>>>(r, r_accum, r_packed, stop);
    c_update_kernel<<<(N_ROWS / 16) / 8 * 32 / 256, 256, 0, stream>>>(
        P, c, r_packed, err_acc, stop, check);
    if (check) err_finalize_kernel<<<1, 1, 0, stream>>>(err_acc, stop);
  }

  output_kernel<<<(size_t)N_ROWS * K_DIM / 4 / 256, 256, 0, stream>>>(P, c, r, out);
}
